// PlainVectorQuantizer_19396072309112
// MI455X (gfx1250) — matrix operand order repaired and checked by reading, not compiled, not run on hardware
//
#include <hip/hip_runtime.h>

// ---------------------------------------------------------------------------
// VQ-VAE vector quantizer for MI455X (gfx1250, wave32, WMMA).
// Distance GEMM done as 3x bf16 WMMA (hi/lo split) ~= fp32 accuracy.
// B-tile staging via GLOBAL_LOAD_ASYNC_TO_LDS_B128 (ASYNCcnt) when available.
// ---------------------------------------------------------------------------

#define D       256
#define K       8192
#define NTOK    16384
#define NUMEL   4194304          // 16*256*32*32
#define LOSS_OFF 4194304
#define IDX_OFF  4194305

typedef __attribute__((ext_vector_type(16))) __bf16 v16bf;
typedef __attribute__((ext_vector_type(8)))  float  v8f;
typedef __attribute__((ext_vector_type(4)))  int    v4i;

#if defined(__gfx1250__) && __has_builtin(__builtin_amdgcn_global_load_async_to_lds_b128)
#define HAVE_ASYNC_LDS 1
#else
#define HAVE_ASYNC_LDS 0
#endif

union Frag {
    uint4 u[2];
    v16bf v;
};

__device__ __forceinline__ unsigned short f32_to_bf16_bits(float x) {
    unsigned int u = __float_as_uint(x);
    u += 0x7FFFu + ((u >> 16) & 1u);          // round-to-nearest-even
    return (unsigned short)(u >> 16);
}

#if HAVE_ASYNC_LDS
// signature (from clang diagnostic): (v4i AS1* src, v4i AS3* dst, imm off, imm cpol)
typedef __attribute__((address_space(1))) v4i* gv4i_p;
typedef __attribute__((address_space(3))) v4i* lv4i_p;

__device__ __forceinline__ void async_b128(void* lds, const void* g) {
    __builtin_amdgcn_global_load_async_to_lds_b128((gv4i_p)g, (lv4i_p)lds, 0, 0);
}
__device__ __forceinline__ void wait_async0() {
#if __has_builtin(__builtin_amdgcn_s_wait_asynccnt)
    __builtin_amdgcn_s_wait_asynccnt(0);
#else
    asm volatile("s_wait_asynccnt 0" ::: "memory");
#endif
}
#endif

// --------------------------- prep: embedding -------------------------------
__global__ void vq_prep_emb(const float* __restrict__ emb,
                            unsigned short* __restrict__ ehi,
                            unsigned short* __restrict__ elo,
                            float* __restrict__ enorm) {
    const int k = blockIdx.x;
    const int d = threadIdx.x;
    float v = emb[(size_t)k * D + d];
    unsigned int u  = __float_as_uint(v);
    unsigned int hb = (u + 0x7FFFu + ((u >> 16) & 1u)) & 0xFFFF0000u;
    ehi[(size_t)k * D + d] = (unsigned short)(hb >> 16);
    elo[(size_t)k * D + d] = f32_to_bf16_bits(v - __uint_as_float(hb));

    float s = v * v;
    #pragma unroll
    for (int off = 16; off > 0; off >>= 1) s += __shfl_xor(s, off, 32);
    __shared__ float part[8];
    if ((threadIdx.x & 31) == 0) part[threadIdx.x >> 5] = s;
    __syncthreads();
    if (threadIdx.x == 0) {
        float t = 0.f;
        #pragma unroll
        for (int i = 0; i < 8; ++i) t += part[i];
        enorm[k] = t;
    }
}

// --------------------------- prep: x transpose -----------------------------
// x[b,d,h,w] f32  ->  z[token][d] bf16 hi/lo, token = b*1024 + h*32 + w
__global__ void vq_prep_x(const float* __restrict__ x,
                          unsigned short* __restrict__ zhi,
                          unsigned short* __restrict__ zlo) {
    __shared__ float tile[32][33];
    const int tx = threadIdx.x, ty = threadIdx.y;
    const int hwt = blockIdx.x, dt = blockIdx.y, b = blockIdx.z;
    const size_t xb = (size_t)b * 262144;
    #pragma unroll
    for (int i = 0; i < 4; ++i) {
        int dloc = ty + i * 8;
        tile[dloc][tx] = x[xb + (size_t)(dt * 32 + dloc) * 1024 + hwt * 32 + tx];
    }
    __syncthreads();
    #pragma unroll
    for (int i = 0; i < 4; ++i) {
        int hwloc = ty + i * 8;
        int token = b * 1024 + hwt * 32 + hwloc;
        float v = tile[tx][hwloc];
        unsigned int u  = __float_as_uint(v);
        unsigned int hb = (u + 0x7FFFu + ((u >> 16) & 1u)) & 0xFFFF0000u;
        size_t o = (size_t)token * D + dt * 32 + tx;
        zhi[o] = (unsigned short)(hb >> 16);
        zlo[o] = f32_to_bf16_bits(v - __uint_as_float(hb));
    }
}

// --------------------------- main: distances + argmin + gather -------------
__global__ __launch_bounds__(256)
void vq_main(const float* __restrict__ x,
             const float* __restrict__ emb,
             const unsigned short* __restrict__ ehi,
             const unsigned short* __restrict__ elo,
             const unsigned short* __restrict__ zhi,
             const unsigned short* __restrict__ zlo,
             const float* __restrict__ enorm,
             float* __restrict__ out,
             float* __restrict__ lossAcc) {
    // B double buffer: 16 emb rows x 256 bf16 (512B) padded to 528B rows
    __shared__ __align__(16) unsigned char ldsB[2][2][16 * 528];
    __shared__ float enormLds[2][16];
    __shared__ int   idxLds[128];

    const int tid  = threadIdx.x;
    const int lane = tid & 31;
    const int wave = tid >> 5;
    const int g    = blockIdx.x;
    const int tokBase = g * 128 + wave * 16;
    const int row = lane & 15;          // M (A) / N (B) index
    const int hf  = lane >> 4;          // half-wave selector

    // ---- A fragments (16 tokens x 256 D, hi+lo) resident in VGPRs ----
    // k order used here (contiguous, same as B): lanes 0-15 hold K 0..15; lanes 16-31 hold K 16..31.
    Frag Ah[8], Al[8];
    {
        const char* zh = (const char*)zhi;
        const char* zl = (const char*)zlo;
        size_t rb = ((size_t)(tokBase + row) * D) * 2 + hf * 32;
        #pragma unroll
        for (int s = 0; s < 8; ++s) {
            size_t o = rb + (size_t)s * 64;
            Ah[s].u[0] = *(const uint4*)(zh + o);
            Ah[s].u[1] = *(const uint4*)(zh + o + 16);
            Al[s].u[0] = *(const uint4*)(zl + o);
            Al[s].u[1] = *(const uint4*)(zl + o + 16);
        }
    }

    const char* ehc = (const char*)ehi;
    const char* elc = (const char*)elo;

    // per-thread staging coordinates (two 16B chunks per matrix per chunk)
    const int sr0 = tid >> 5;                 // row of first 16B chunk
    const int sc0 = (tid & 31) << 4;          // byte col of first chunk
    const int sr1 = (256 + tid) >> 5;
    const int sc1 = ((256 + tid) & 31) << 4;

    float minv[8];
    int   mini[8];
    #pragma unroll
    for (int r = 0; r < 8; ++r) { minv[r] = 3.4e38f; mini[r] = 0; }

#if HAVE_ASYNC_LDS
    // ---- stage chunk 0 into buffer 0 via async DMA (ASYNCcnt) ----
    async_b128(&ldsB[0][0][sr0 * 528 + sc0], ehc + (size_t)sr0 * 512 + sc0);
    async_b128(&ldsB[0][0][sr1 * 528 + sc1], ehc + (size_t)sr1 * 512 + sc1);
    async_b128(&ldsB[0][1][sr0 * 528 + sc0], elc + (size_t)sr0 * 512 + sc0);
    async_b128(&ldsB[0][1][sr1 * 528 + sc1], elc + (size_t)sr1 * 512 + sc1);
    if (tid < 16) enormLds[0][tid] = enorm[tid];
    wait_async0();
    __syncthreads();

    for (int c = 0; c < 512; ++c) {
        const int buf = c & 1;
        const int nb  = buf ^ 1;

        // fire-and-forget DMA of next chunk; completes under the WMMAs below
        if (c < 511) {
            size_t sb = (size_t)(c + 1) * 16 * 512;
            async_b128(&ldsB[nb][0][sr0 * 528 + sc0], ehc + sb + (size_t)sr0 * 512 + sc0);
            async_b128(&ldsB[nb][0][sr1 * 528 + sc1], ehc + sb + (size_t)sr1 * 512 + sc1);
            async_b128(&ldsB[nb][1][sr0 * 528 + sc0], elc + sb + (size_t)sr0 * 512 + sc0);
            async_b128(&ldsB[nb][1][sr1 * 528 + sc1], elc + sb + (size_t)sr1 * 512 + sc1);
            if (tid < 16) enormLds[nb][tid] = enorm[(c + 1) * 16 + tid];
        }

        // ---- 16x16 distance tile: 8 k-steps x 3 WMMA (hh, hl, lh) ----
        v8f acc_hh = {}, acc_hl = {}, acc_lh = {};
        const unsigned char* bh = ldsB[buf][0];
        const unsigned char* bl = ldsB[buf][1];
        const int fo = row * 528 + hf * 32;   // B: lane=N, K consecutive/half
        #pragma unroll
        for (int s = 0; s < 8; ++s) {
            Frag Bh, Bl;
            const uint4* ph = (const uint4*)(bh + fo + s * 64);
            const uint4* pl = (const uint4*)(bl + fo + s * 64);
            Bh.u[0] = ph[0]; Bh.u[1] = ph[1];
            Bl.u[0] = pl[0]; Bl.u[1] = pl[1];
            acc_hh = __builtin_amdgcn_wmma_f32_16x16x32_bf16(
                false, Ah[s].v, false, Bh.v, (short)0, acc_hh, false, false);
            acc_hl = __builtin_amdgcn_wmma_f32_16x16x32_bf16(
                false, Ah[s].v, false, Bl.v, (short)0, acc_hl, false, false);
            acc_lh = __builtin_amdgcn_wmma_f32_16x16x32_bf16(
                false, Al[s].v, false, Bh.v, (short)0, acc_lh, false, false);
        }

        // score = ||e||^2 - 2 z.e  (||z||^2 constant under argmin)
        const float en  = enormLds[buf][row];
        const int   col = c * 16 + row;
        #pragma unroll
        for (int r = 0; r < 8; ++r) {
            float dot = acc_hh[r] + acc_hl[r] + acc_lh[r];
            float sc  = fmaf(-2.f, dot, en);
            if (sc < minv[r]) { minv[r] = sc; mini[r] = col; }
        }

        if (c < 511) wait_async0();   // this wave's DMA into buf^1 landed
        __syncthreads();              // all waves done reading buf / writing nb
    }
#else
    // ---- fallback: register-bounce staging (global->VGPR->ds_store) ----
    #pragma unroll
    for (int p = 0; p < 2; ++p) {
        int e  = p * 256 + tid;
        int r  = e >> 5;
        int cb = (e & 31) << 4;
        *(uint4*)(&ldsB[0][0][r * 528 + cb]) = *(const uint4*)(ehc + (size_t)r * 512 + cb);
        *(uint4*)(&ldsB[0][1][r * 528 + cb]) = *(const uint4*)(elc + (size_t)r * 512 + cb);
    }
    if (tid < 16) enormLds[0][tid] = enorm[tid];
    __syncthreads();

    for (int c = 0; c < 512; ++c) {
        const int buf = c & 1;
        uint4 sh0, sh1, sl0, sl1;
        float enNext = 0.f;
        if (c < 511) {
            size_t sb = (size_t)(c + 1) * 16 * 512;
            sh0 = *(const uint4*)(ehc + sb + (size_t)sr0 * 512 + sc0);
            sh1 = *(const uint4*)(ehc + sb + (size_t)sr1 * 512 + sc1);
            sl0 = *(const uint4*)(elc + sb + (size_t)sr0 * 512 + sc0);
            sl1 = *(const uint4*)(elc + sb + (size_t)sr1 * 512 + sc1);
            if (tid < 16) enNext = enorm[(c + 1) * 16 + tid];
        }

        v8f acc_hh = {}, acc_hl = {}, acc_lh = {};
        const unsigned char* bh = ldsB[buf][0];
        const unsigned char* bl = ldsB[buf][1];
        const int fo = row * 528 + hf * 32;
        #pragma unroll
        for (int s = 0; s < 8; ++s) {
            Frag Bh, Bl;
            const uint4* ph = (const uint4*)(bh + fo + s * 64);
            const uint4* pl = (const uint4*)(bl + fo + s * 64);
            Bh.u[0] = ph[0]; Bh.u[1] = ph[1];
            Bl.u[0] = pl[0]; Bl.u[1] = pl[1];
            acc_hh = __builtin_amdgcn_wmma_f32_16x16x32_bf16(
                false, Ah[s].v, false, Bh.v, (short)0, acc_hh, false, false);
            acc_hl = __builtin_amdgcn_wmma_f32_16x16x32_bf16(
                false, Ah[s].v, false, Bl.v, (short)0, acc_hl, false, false);
            acc_lh = __builtin_amdgcn_wmma_f32_16x16x32_bf16(
                false, Al[s].v, false, Bh.v, (short)0, acc_lh, false, false);
        }

        const float en  = enormLds[buf][row];
        const int   col = c * 16 + row;
        #pragma unroll
        for (int r = 0; r < 8; ++r) {
            float dot = acc_hh[r] + acc_hl[r] + acc_lh[r];
            float sc  = fmaf(-2.f, dot, en);
            if (sc < minv[r]) { minv[r] = sc; mini[r] = col; }
        }

        __syncthreads();
        if (c < 511) {
            const int nb = buf ^ 1;
            *(uint4*)(&ldsB[nb][0][sr0 * 528 + sc0]) = sh0;
            *(uint4*)(&ldsB[nb][0][sr1 * 528 + sc1]) = sh1;
            *(uint4*)(&ldsB[nb][1][sr0 * 528 + sc0]) = sl0;
            *(uint4*)(&ldsB[nb][1][sr1 * 528 + sc1]) = sl1;
            if (tid < 16) enormLds[nb][tid] = enNext;
            __syncthreads();
        }
    }
#endif

    // ---- cross-lane argmin: C layout rows r (lanes 0-15) / r+8 (16-31) ----
    #pragma unroll
    for (int r = 0; r < 8; ++r) {
        float v = minv[r];
        int   i = mini[r];
        #pragma unroll
        for (int off = 8; off > 0; off >>= 1) {
            float v2 = __shfl_xor(v, off, 32);
            int   i2 = __shfl_xor(i, off, 32);
            if (v2 < v || (v2 == v && i2 < i)) { v = v2; i = i2; }
        }
        if (row == 0) {
            int m = r + hf * 8;
            idxLds[wave * 16 + m] = i;
            out[IDX_OFF + tokBase + m] = (float)i;   // indices as f32
        }
    }
    __syncthreads();

    // ---- gather codebook rows, write BCHW output, accumulate loss ----
    const int tloc  = tid & 127;
    const int token = g * 128 + tloc;
    const int b     = token >> 10;
    const int hw    = token & 1023;
    const int ei    = idxLds[tloc];
    const float* erow = emb + (size_t)ei * D;
    const size_t obase = (size_t)b * 262144 + hw;
    float lsum = 0.f;
    for (int d = (tid >> 7); d < D; d += 2) {
        float  q  = erow[d];
        size_t o  = obase + (size_t)d * 1024;
        float  zv = x[o];                       // exact f32 z for the loss
        out[o] = q;                             // straight-through value == q
        float df = q - zv;
        lsum = fmaf(df, df, lsum);
    }
    #pragma unroll
    for (int off = 16; off > 0; off >>= 1) lsum += __shfl_xor(lsum, off, 32);
    if (lane == 0) atomicAdd(lossAcc, lsum);
}

// --------------------------- init / finalize -------------------------------
__global__ void vq_init(float* lossAcc) {
    if (threadIdx.x == 0) *lossAcc = 0.f;
}

__global__ void vq_finalize(const float* __restrict__ lossAcc,
                            float* __restrict__ out) {
    // loss = q_latent + 0.25 * e_latent = 1.25 * mean((q - z)^2)
    out[LOSS_OFF] = 1.25f * (*lossAcc) / (float)NUMEL;
}

// ---------------------------------------------------------------------------
extern "C" void kernel_launch(void* const* d_in, const int* in_sizes, int n_in,
                              void* d_out, int out_size, void* d_ws, size_t ws_size,
                              hipStream_t stream) {
    (void)in_sizes; (void)n_in; (void)out_size; (void)ws_size;
    const float* x   = (const float*)d_in[0];
    const float* emb = (const float*)d_in[1];
    float* out = (float*)d_out;

    char* W = (char*)d_ws;
    float* lossAcc = (float*)W;                         // 4 B
    float* enorm   = (float*)(W + 256);                 // 32 KB
    unsigned short* ehi = (unsigned short*)(W + 65536); // 4 MB
    unsigned short* elo = ehi + (size_t)K * D;          // 4 MB
    unsigned short* zhi = elo + (size_t)K * D;          // 8 MB
    unsigned short* zlo = zhi + (size_t)NTOK * D;       // 8 MB

    vq_init<<<1, 32, 0, stream>>>(lossAcc);
    vq_prep_emb<<<K, 256, 0, stream>>>(emb, ehi, elo, enorm);
    vq_prep_x<<<dim3(32, 8, 16), dim3(32, 8), 0, stream>>>(x, zhi, zlo);
    vq_main<<<128, 256, 0, stream>>>(x, emb, ehi, elo, zhi, zlo, enorm, out, lossAcc);
    vq_finalize<<<1, 1, 0, stream>>>(lossAcc, out);
}
